// SwiGLUAttention_5944234737727
// MI455X (gfx1250) — matrix operand order repaired and checked by reading, not compiled, not run on hardware
//
#include <hip/hip_runtime.h>
#include <cstdint>

#define DIM    1024
#define NHEAD  16
#define HDIM   64
#define HIDDEN 2048
#define BATCH  4
#define SEQ    2048
#define MTOT   (BATCH*SEQ)   // 8192 rows

typedef __attribute__((ext_vector_type(16))) __bf16 v16bf;
typedef __attribute__((ext_vector_type(8)))  float  v8f;
typedef __attribute__((ext_vector_type(4)))  float  f32x4;
typedef __attribute__((ext_vector_type(4)))  unsigned int u32x4;
typedef __attribute__((ext_vector_type(8)))  unsigned int u32x8;

union FragU { v16bf v; f32x4 f[2]; };

// A-fragment (16x32 bf16): lane -> row = lane&15; elems 0..7 -> K=g*8+j, 8..15 -> K=16+g*8+j
__device__ inline v16bf frag_a(const __bf16* base, int stride, int lane) {
  const int r = lane & 15, g = lane >> 4;
  const __bf16* p = base + r * stride + g * 8;
  FragU u;
  u.f[0] = *reinterpret_cast<const f32x4*>(p);
  u.f[1] = *reinterpret_cast<const f32x4*>(p + 16);
  return u.v;
}
// B-fragment (32x16 bf16): lane -> col = lane&15; elems 0..7 -> K=g*8+j, 8..15 -> K=16+g*8+j
__device__ inline v16bf frag_b(const __bf16* base, int stride, int lane) {
  const int n = lane & 15, g = lane >> 4;
  const __bf16* p = base + n * stride + g * 8;
  FragU u;
  u.f[0] = *reinterpret_cast<const f32x4*>(p);
  u.f[1] = *reinterpret_cast<const f32x4*>(p + 16);
  return u.v;
}

__device__ inline void wait_tensorcnt0() {
#if __has_builtin(__builtin_amdgcn_s_wait_tensorcnt)
  __builtin_amdgcn_s_wait_tensorcnt((short)0);
#else
  asm volatile("s_wait_tensorcnt 0x0" ::: "memory");
#endif
}

// One 2D TDM DMA: global (row-major, stride0 elements of 2B) -> LDS, compact
// tile_d1 x tile_d0 tile.  D# built per CDNA5 ISA.
__device__ inline void tdm_load_2d(unsigned lds_addr, const __bf16* gptr,
                                   unsigned tensor_d0, unsigned tensor_d1,
                                   unsigned tile_d0, unsigned tile_d1,
                                   unsigned stride0) {
  const unsigned long long ga = (unsigned long long)(uintptr_t)gptr;
  u32x4 g0;
  g0[0] = 1u;                                   // count=1, user, no gather
  g0[1] = lds_addr;                             // LDS byte address
  g0[2] = (unsigned)(ga & 0xffffffffu);         // global_addr[31:0]
  g0[3] = (unsigned)((ga >> 32) & 0x01ffffffu)  // global_addr[56:32]
          | 0x80000000u;                        // type=2 ("image") in [127:126]
  u32x8 g1;
  g1[0] = (1u << 16);                           // data_size=1 -> 2 bytes
  g1[1] = (tensor_d0 & 0xffffu) << 16;          // tensor_dim0[15:0]
  g1[2] = (tensor_d0 >> 16) | ((tensor_d1 & 0xffffu) << 16);
  g1[3] = (tensor_d1 >> 16) | (tile_d0 << 16);  // tile_dim0
  g1[4] = tile_d1;                              // tile_dim1, tile_dim2=0
  g1[5] = stride0;                              // tensor_dim0_stride[31:0]
  g1[6] = 0u;
  g1[7] = 0u;
  asm volatile("tensor_load_to_lds %0, %1" :: "s"(g0), "s"(g1) : "memory");
}

// ---------------------------------------------------------------------------
// GEMM: C[M,F] = A[M,K](bf16,rm) * W[F,K]^T(bf16,rm) + bias.
// Block tile 128x256, 8 waves each computing 64x64 (4x4 WMMA tiles),
// K-step 32.  Tiles staged by TDM, double-buffered: 16 WMMA per 16
// ds_load_b128 per wave per K-step.
// ---------------------------------------------------------------------------
#define A_TILE_ELEMS (128 * 32)
#define W_TILE_ELEMS (256 * 32)
template <typename OutT>
__global__ __launch_bounds__(256) void gemm_bf16_wmma(
    const __bf16* __restrict__ A, const __bf16* __restrict__ W,
    const float* __restrict__ bias, OutT* __restrict__ C,
    int M, int F, int K)
{
  __shared__ __align__(16) __bf16 As[2][A_TILE_ELEMS];   // 16 KB
  __shared__ __align__(16) __bf16 Ws[2][W_TILE_ELEMS];   // 32 KB

  const int tid  = threadIdx.x;
  const int lane = tid & 31, wid = tid >> 5;
  const int wm = wid & 1;          // 2 waves along M -> 64 rows
  const int wn = wid >> 1;         // 4 waves along F -> 64 cols
  const int m0 = blockIdx.y * 128, f0 = blockIdx.x * 256;
  const bool issuer = (wid == 0);

  v8f acc[4][4];
  #pragma unroll
  for (int i = 0; i < 4; i++)
    #pragma unroll
    for (int j = 0; j < 4; j++)
      #pragma unroll
      for (int r = 0; r < 8; r++) acc[i][j][r] = 0.f;

  const unsigned ldsA = (unsigned)(uintptr_t)&As[0][0];
  const unsigned ldsW = (unsigned)(uintptr_t)&Ws[0][0];

  if (issuer) {   // prime buffer 0
    tdm_load_2d(ldsA, A + (size_t)m0 * K, (unsigned)K, (unsigned)(M - m0),
                32u, 128u, (unsigned)K);
    tdm_load_2d(ldsW, W + (size_t)f0 * K, (unsigned)K, (unsigned)(F - f0),
                32u, 256u, (unsigned)K);
  }

  int buf = 0;
  for (int k0 = 0; k0 < K; k0 += 32, buf ^= 1) {
    if (issuer) wait_tensorcnt0();
    __syncthreads();                       // buffer `buf` ready for everyone
    if (issuer && (k0 + 32 < K)) {         // overlap: DMA next tile into buf^1
      const int kn = k0 + 32;
      tdm_load_2d(ldsA + (unsigned)((buf ^ 1) * A_TILE_ELEMS * 2),
                  A + (size_t)m0 * K + kn, (unsigned)(K - kn),
                  (unsigned)(M - m0), 32u, 128u, (unsigned)K);
      tdm_load_2d(ldsW + (unsigned)((buf ^ 1) * W_TILE_ELEMS * 2),
                  W + (size_t)f0 * K + kn, (unsigned)(K - kn),
                  (unsigned)(F - f0), 32u, 256u, (unsigned)K);
    }
    const __bf16* Ab = As[buf];
    const __bf16* Wb = Ws[buf];
    v16bf bfrag[4];
    #pragma unroll
    for (int j = 0; j < 4; j++)
      bfrag[j] = frag_b(&Wb[(wn * 64 + j * 16) * 32], 32, lane);
    #pragma unroll
    for (int i = 0; i < 4; i++) {
      v16bf af = frag_a(&Ab[(wm * 64 + i * 16) * 32], 32, lane);
      #pragma unroll
      for (int j = 0; j < 4; j++)
        acc[i][j] = __builtin_amdgcn_wmma_f32_16x16x32_bf16(
            false, af, false, bfrag[j], (short)0, acc[i][j], false, false);
    }
    // no trailing barrier: next iteration's wait+barrier guards buffer reuse
  }

  const int cn = lane & 15, g = lane >> 4;
  #pragma unroll
  for (int i = 0; i < 4; i++)
    #pragma unroll
    for (int j = 0; j < 4; j++) {
      const int col = f0 + wn * 64 + j * 16 + cn;
      const float bv = bias ? bias[col] : 0.f;
      #pragma unroll
      for (int r = 0; r < 8; r++) {
        const int row = m0 + wm * 64 + i * 16 + g * 8 + r;
        C[(size_t)row * F + col] = (OutT)(acc[i][j][r] + bv);
      }
    }
}

// ---------------------------------------------------------------------------
__global__ void f32_to_bf16(const float* __restrict__ in, __bf16* __restrict__ out, int n) {
  int i = blockIdx.x * blockDim.x + threadIdx.x;
  if (i < n) out[i] = (__bf16)in[i];
}

// RoPE + split: qkv[B,N,3,H,D] -> q (rotated, pre-scaled by 1/sqrt(D)), k
// (rotated), v in [B,H,N,D]
__global__ void rope_split(const __bf16* __restrict__ qkv,
                           __bf16* __restrict__ Qo, __bf16* __restrict__ Ko,
                           __bf16* __restrict__ Vo, int N)
{
  int idx = blockIdx.x * blockDim.x + threadIdx.x;
  if (idx >= BATCH * NHEAD * N * 32) return;
  const int d = idx & 31;
  int t = idx >> 5;
  const int n = t % N; t /= N;
  const int h = t % NHEAD;
  const int b = t / NHEAD;

  const __bf16* qp = qkv + ((size_t)(b * (size_t)N + n)) * 3072 + h * 64;
  const float q0 = (float)qp[d],        q1 = (float)qp[d + 32];
  const float k0 = (float)qp[1024 + d], k1 = (float)qp[1024 + d + 32];
  const float v0 = (float)qp[2048 + d], v1 = (float)qp[2048 + d + 32];

  const float inv = __expf(-(float)d * (9.210340371976184f / 32.f)); // 10000^(-d/32)
  float sn, cs;
  __sincosf((float)n * inv, &sn, &cs);

  const size_t ob = ((size_t)(b * NHEAD + h) * N + n) * 64;
  const float sc = 0.125f;  // HEAD_DIM^-0.5 folded into Q
  Qo[ob + d]      = (__bf16)((q0 * cs - q1 * sn) * sc);
  Qo[ob + d + 32] = (__bf16)((q1 * cs + q0 * sn) * sc);
  Ko[ob + d]      = (__bf16)(k0 * cs - k1 * sn);
  Ko[ob + d + 32] = (__bf16)(k1 * cs + k0 * sn);
  Vo[ob + d]      = (__bf16)v0;
  Vo[ob + d + 32] = (__bf16)v1;
}

// width-16 XOR reductions via single ds_swizzle (and=0x1f, or=0, xor=MASK).
template <int MASK>
__device__ inline float swz_f(float x) {
  return __int_as_float(__builtin_amdgcn_ds_swizzle(__float_as_int(x), 0x7c00 | MASK));
}
__device__ inline float red16_max(float x) {
  x = fmaxf(x, swz_f<1>(x));
  x = fmaxf(x, swz_f<2>(x));
  x = fmaxf(x, swz_f<4>(x));
  x = fmaxf(x, swz_f<8>(x));
  return x;
}
__device__ inline float red16_sum(float x) {
  x += swz_f<1>(x);
  x += swz_f<2>(x);
  x += swz_f<4>(x);
  x += swz_f<8>(x);
  return x;
}

// ---------------------------------------------------------------------------
// Flash attention (non-causal): 4 waves x 16 query rows per block, 64-key
// stream with online softmax; WMMA for QK^T and PV.
// ---------------------------------------------------------------------------
__global__ __launch_bounds__(128) void flash_attn_wmma(
    const __bf16* __restrict__ Qg, const __bf16* __restrict__ Kg,
    const __bf16* __restrict__ Vg, __bf16* __restrict__ Og, int N)
{
  __shared__ __align__(16) __bf16 Qs[64 * 64];
  __shared__ __align__(16) __bf16 Ks[64 * 64];
  __shared__ __align__(16) __bf16 Vt[64 * 64];      // transposed: [d][key]
  __shared__ __align__(16) __bf16 Ps[4 * 16 * 64];  // per-wave P tile

  const int tid = threadIdx.x, lane = tid & 31, w = tid >> 5;
  const int bh = blockIdx.y;
  const int n0 = blockIdx.x * 64;
  const size_t base = (size_t)bh * N * 64;

  {
    const int row = tid >> 1, half = tid & 1;
    const f32x4* s = reinterpret_cast<const f32x4*>(Qg + base + (size_t)(n0 + row) * 64 + half * 32);
    f32x4* d = reinterpret_cast<f32x4*>(&Qs[row * 64 + half * 32]);
    d[0] = s[0]; d[1] = s[1]; d[2] = s[2]; d[3] = s[3];
  }

  v8f O[4];
  float mrow[8], lrow[8];
  #pragma unroll
  for (int t = 0; t < 4; t++)
    #pragma unroll
    for (int r = 0; r < 8; r++) O[t][r] = 0.f;
  #pragma unroll
  for (int r = 0; r < 8; r++) { mrow[r] = -3.0e38f; lrow[r] = 0.f; }

  const int cn = lane & 15, g = lane >> 4;
  __bf16* Pw = &Ps[w * 16 * 64];

  for (int kb = 0; kb < N; kb += 64) {
    {
      const int row = tid >> 1, half = tid & 1;
      const f32x4* s = reinterpret_cast<const f32x4*>(Kg + base + (size_t)(kb + row) * 64 + half * 32);
      f32x4* d = reinterpret_cast<f32x4*>(&Ks[row * 64 + half * 32]);
      d[0] = s[0]; d[1] = s[1]; d[2] = s[2]; d[3] = s[3];
      const __bf16* vs = Vg + base + (size_t)(kb + row) * 64 + half * 32;
      #pragma unroll
      for (int dd = 0; dd < 32; dd++) Vt[(half * 32 + dd) * 64 + row] = vs[dd];
    }
    __syncthreads();

    // S = Q K^T (Q pre-scaled): 16 x 64 per wave
    v8f S[4];
    #pragma unroll
    for (int ct = 0; ct < 4; ct++) {
      #pragma unroll
      for (int r = 0; r < 8; r++) S[ct][r] = 0.f;
      #pragma unroll
      for (int ks = 0; ks < 2; ks++) {
        v16bf a = frag_a(&Qs[(w * 16) * 64 + ks * 32], 64, lane);
        v16bf b = frag_b(&Ks[(ct * 16) * 64 + ks * 32], 64, lane);
        S[ct] = __builtin_amdgcn_wmma_f32_16x16x32_bf16(
            false, a, false, b, (short)0, S[ct], false, false);
      }
    }

    // Online softmax, fully unrolled (rows r+g*8 live across 16 lanes/half)
    #pragma unroll
    for (int r = 0; r < 8; r++) {
      float mx = fmaxf(fmaxf(S[0][r], S[1][r]), fmaxf(S[2][r], S[3][r]));
      mx = red16_max(mx);
      const float mnew = fmaxf(mrow[r], mx);
      const float corr = __expf(mrow[r] - mnew);
      mrow[r] = mnew;
      const float p0 = __expf(S[0][r] - mnew);
      const float p1 = __expf(S[1][r] - mnew);
      const float p2 = __expf(S[2][r] - mnew);
      const float p3 = __expf(S[3][r] - mnew);
      Pw[(g * 8 + r) * 64 +  0 + cn] = (__bf16)p0;
      Pw[(g * 8 + r) * 64 + 16 + cn] = (__bf16)p1;
      Pw[(g * 8 + r) * 64 + 32 + cn] = (__bf16)p2;
      Pw[(g * 8 + r) * 64 + 48 + cn] = (__bf16)p3;
      const float rs = red16_sum((p0 + p1) + (p2 + p3));
      lrow[r] = lrow[r] * corr + rs;
      O[0][r] *= corr; O[1][r] *= corr; O[2][r] *= corr; O[3][r] *= corr;
    }
    asm volatile("s_wait_dscnt 0" ::: "memory");  // same-wave cross-lane P RAW

    // O += P V
    #pragma unroll
    for (int kc = 0; kc < 2; kc++) {
      v16bf a = frag_a(&Pw[kc * 32], 64, lane);
      #pragma unroll
      for (int dt = 0; dt < 4; dt++) {
        v16bf b = frag_b(&Vt[(dt * 16) * 64 + kc * 32], 64, lane);
        O[dt] = __builtin_amdgcn_wmma_f32_16x16x32_bf16(
            false, a, false, b, (short)0, O[dt], false, false);
      }
    }
    __syncthreads();
  }

  const int b = bh >> 4, h = bh & 15;
  #pragma unroll
  for (int dt = 0; dt < 4; dt++) {
    #pragma unroll
    for (int r = 0; r < 8; r++) {
      const int n = n0 + w * 16 + g * 8 + r;
      const float v = O[dt][r] / lrow[r];
      Og[((size_t)b * N + n) * DIM + h * 64 + dt * 16 + cn] = (__bf16)v;
    }
  }
}

// ---------------------------------------------------------------------------
__global__ void swiglu_ew(const __bf16* __restrict__ x1, const __bf16* __restrict__ x2,
                          __bf16* __restrict__ o, int n) {
  int i = blockIdx.x * blockDim.x + threadIdx.x;
  if (i < n) {
    const float a = (float)x1[i];
    const float s = a / (1.f + __expf(-a));
    o[i] = (__bf16)(s * (float)x2[i]);
  }
}

// ---------------------------------------------------------------------------
extern "C" void kernel_launch(void* const* d_in, const int* in_sizes, int n_in,
                              void* d_out, int out_size, void* d_ws, size_t ws_size,
                              hipStream_t stream)
{
  const float* x      = (const float*)d_in[0];
  const float* qkv_w  = (const float*)d_in[1];
  const float* proj_w = (const float*)d_in[2];
  const float* proj_b = (const float*)d_in[3];
  const float* w1_w   = (const float*)d_in[4];
  const float* w1_b   = (const float*)d_in[5];
  const float* w2_w   = (const float*)d_in[6];
  const float* w2_b   = (const float*)d_in[7];
  const float* w3_w   = (const float*)d_in[8];
  const float* w3_b   = (const float*)d_in[9];
  float* out = (float*)d_out;

  char* ws = (char*)d_ws;
  size_t off = 0;
  auto alloc = [&](size_t bytes) -> __bf16* {
    __bf16* p = (__bf16*)(ws + off);
    off += (bytes + 255) & ~(size_t)255;
    return p;
  };
  __bf16* x_bf    = alloc((size_t)MTOT * DIM * 2);
  __bf16* wq_bf   = alloc((size_t)3 * DIM * DIM * 2);
  __bf16* wp_bf   = alloc((size_t)DIM * DIM * 2);
  __bf16* w1_bf   = alloc((size_t)HIDDEN * DIM * 2);
  __bf16* w2_bf   = alloc((size_t)HIDDEN * DIM * 2);
  __bf16* w3_bf   = alloc((size_t)DIM * HIDDEN * 2);
  __bf16* qkv_bf  = alloc((size_t)MTOT * 3 * DIM * 2);   // later aliased: x1
  __bf16* q_bf    = alloc((size_t)MTOT * DIM * 2);       // q..k later: x2
  __bf16* k_bf    = alloc((size_t)MTOT * DIM * 2);
  __bf16* v_bf    = alloc((size_t)MTOT * DIM * 2);       // v..attn later: hidden
  __bf16* attn_bf = alloc((size_t)MTOT * DIM * 2);
  __bf16* proj_bf = alloc((size_t)MTOT * DIM * 2);
  __bf16* x1_bf  = qkv_bf;
  __bf16* x2_bf  = q_bf;
  __bf16* hid_bf = v_bf;
  (void)ws_size; (void)in_sizes; (void)n_in; (void)out_size;

  const int THR = 256;
  auto cvt = [&](const float* s, __bf16* d, size_t n) {
    f32_to_bf16<<<dim3((unsigned)((n + THR - 1) / THR)), dim3(THR), 0, stream>>>(s, d, (int)n);
  };
  cvt(x,      x_bf,  (size_t)MTOT * DIM);
  cvt(qkv_w,  wq_bf, (size_t)3 * DIM * DIM);
  cvt(proj_w, wp_bf, (size_t)DIM * DIM);
  cvt(w1_w,   w1_bf, (size_t)HIDDEN * DIM);
  cvt(w2_w,   w2_bf, (size_t)HIDDEN * DIM);
  cvt(w3_w,   w3_bf, (size_t)DIM * HIDDEN);

  gemm_bf16_wmma<__bf16><<<dim3(3 * DIM / 256, MTOT / 128), 256, 0, stream>>>(
      x_bf, wq_bf, nullptr, qkv_bf, MTOT, 3 * DIM, DIM);

  {
    const size_t n = (size_t)BATCH * NHEAD * SEQ * 32;
    rope_split<<<dim3((unsigned)((n + THR - 1) / THR)), THR, 0, stream>>>(
        qkv_bf, q_bf, k_bf, v_bf, SEQ);
  }

  flash_attn_wmma<<<dim3(SEQ / 64, BATCH * NHEAD), 128, 0, stream>>>(
      q_bf, k_bf, v_bf, attn_bf, SEQ);

  gemm_bf16_wmma<__bf16><<<dim3(DIM / 256, MTOT / 128), 256, 0, stream>>>(
      attn_bf, wp_bf, proj_b, proj_bf, MTOT, DIM, DIM);

  gemm_bf16_wmma<__bf16><<<dim3(HIDDEN / 256, MTOT / 128), 256, 0, stream>>>(
      proj_bf, w1_bf, w1_b, x1_bf, MTOT, HIDDEN, DIM);
  gemm_bf16_wmma<__bf16><<<dim3(HIDDEN / 256, MTOT / 128), 256, 0, stream>>>(
      proj_bf, w2_bf, w2_b, x2_bf, MTOT, HIDDEN, DIM);
  {
    const int n = MTOT * HIDDEN;
    swiglu_ew<<<dim3((n + THR - 1) / THR), THR, 0, stream>>>(x1_bf, x2_bf, hid_bf, n);
  }

  gemm_bf16_wmma<float><<<dim3(DIM / 256, MTOT / 128), 256, 0, stream>>>(
      hid_bf, w3_bf, w3_b, out, MTOT, DIM, HIDDEN);
}
